// PolarizedMamba2Block_55843164783329
// MI455X (gfx1250) — matrix operand order repaired and checked by reading, not compiled, not run on hardware
//
#include <hip/hip_runtime.h>
#include <hip/hip_bf16.h>

typedef _Float16 half_t;
typedef __attribute__((ext_vector_type(16))) _Float16 v16h;
typedef __attribute__((ext_vector_type(8)))  _Float16 v8h;
typedef __attribute__((ext_vector_type(8)))  float    v8f;

// Problem constants (from reference)
#define BB      2
#define LL      4096
#define DD      1024
#define DIN     2048
#define DSTATE  128
#define DCONV   4
#define HDIM    64
#define NH      32
#define CONVDIM 2304          // DIN + 2*DSTATE
#define DPROJ   4384          // 2*DIN + 2*DSTATE + NH
#define DFUSE   6144          // 3*DIN
#define MROWS   (BB*LL)       // 8192
#define EPS     1e-5f

__device__ __forceinline__ float silu_f(float x)      { return x / (1.0f + __expf(-x)); }
__device__ __forceinline__ float softplus_f(float x)  { return (x > 20.0f) ? x : log1pf(__expf(x)); }

// ---------------------------------------------------------------------------
// Weight transpose + fp32->f16:  W[K][N] (row major) -> Wt[N][K]
// ---------------------------------------------------------------------------
__global__ void k_transpose_f16(const float* __restrict__ W, half_t* __restrict__ Wt,
                                int K, int N) {
    size_t idx = (size_t)blockIdx.x * blockDim.x + threadIdx.x;
    size_t total = (size_t)K * N;
    if (idx >= total) return;
    int k = (int)(idx / N);
    int n = (int)(idx % N);
    Wt[(size_t)n * K + k] = (half_t)W[idx];
}

// ---------------------------------------------------------------------------
// RMSNorm over d=1024: one block per row, 256 threads x 4 elems
// ---------------------------------------------------------------------------
__global__ void k_rmsnorm(const float* __restrict__ x, const float* __restrict__ w,
                          float* __restrict__ xn, half_t* __restrict__ xnh) {
    __shared__ float red[256];
    int row = blockIdx.x;
    int tid = threadIdx.x;
    const float* xp = x + (size_t)row * DD;
    float v[4];
    float ss = 0.f;
#pragma unroll
    for (int j = 0; j < 4; ++j) {
        v[j] = xp[tid + j * 256];
        ss += v[j] * v[j];
    }
    red[tid] = ss;
    __syncthreads();
    for (int s = 128; s > 0; s >>= 1) {
        if (tid < s) red[tid] += red[tid + s];
        __syncthreads();
    }
    float scale = rsqrtf(red[0] * (1.0f / DD) + EPS);
#pragma unroll
    for (int j = 0; j < 4; ++j) {
        int c = tid + j * 256;
        float o = v[j] * scale * w[c];
        xn[(size_t)row * DD + c]  = o;
        xnh[(size_t)row * DD + c] = (half_t)o;
    }
}

// ---------------------------------------------------------------------------
// WMMA GEMM:  C[M,N](f32) = A[M,K](f16,row) * Bt[N,K](f16,row)  (+ addsrc)
// 256 threads = 8 waves; block tile 128x128; wave tile 32x64 (2x4 WMMA tiles)
// Residual add is compile-time specialized to keep the epilogue branch-free.
// ---------------------------------------------------------------------------
template <bool HASADD>
__global__ __launch_bounds__(256)
void k_gemm_wmma(const half_t* __restrict__ A, const half_t* __restrict__ Bt,
                 float* __restrict__ C, const float* __restrict__ addsrc,
                 int M, int N, int K) {
    const int lane    = threadIdx.x & 31;
    const int wid     = threadIdx.x >> 5;
    const int half_id = lane >> 4;        // which 16-lane half of the wave
    const int l16     = lane & 15;
    const int baseRow = blockIdx.y * 128 + (wid & 3) * 32;
    const int baseCol = blockIdx.x * 128 + (wid >> 2) * 64;

    // Per-wave fragment base pointers (row/col fixed across the K loop)
    const half_t* aptr[2];
#pragma unroll
    for (int mi = 0; mi < 2; ++mi)
        aptr[mi] = A + (size_t)(baseRow + mi * 16 + l16) * K + half_id * 16;

    const half_t* bptr[4];
#pragma unroll
    for (int ni = 0; ni < 4; ++ni) {
        int col = baseCol + ni * 16 + l16;
        if (col >= N) col = N - 1;          // clamped lanes only feed dropped outputs
        bptr[ni] = Bt + (size_t)col * K + half_id * 16;
    }

    v8f acc[2][4];
#pragma unroll
    for (int mi = 0; mi < 2; ++mi)
#pragma unroll
        for (int ni = 0; ni < 4; ++ni)
#pragma unroll
            for (int r = 0; r < 8; ++r)
                acc[mi][ni][r] = 0.0f;

    for (int k0 = 0; k0 < K; k0 += 32) {
        // Prefetch 4 K-steps ahead (lowers to global_prefetch_b8; no counter)
        if (k0 + 128 < K) {
            __builtin_prefetch(aptr[0] + k0 + 128, 0, 0);
            __builtin_prefetch(bptr[0] + k0 + 128, 0, 0);
            __builtin_prefetch(bptr[2] + k0 + 128, 0, 0);
        }
        // ---- A fragments: 16-bit A 16x32 layout ----
        // lane (l16, half_id) holds row=baseRow+mi*16+l16,
        // K = half*16 .. half*16+15 (contiguous, matches B)
        v16h afrag[2];
#pragma unroll
        for (int mi = 0; mi < 2; ++mi) {
            const half_t* ap = aptr[mi] + k0;
            v8h lo = *(const v8h*)(ap);
            v8h hi = *(const v8h*)(ap + 8);
#pragma unroll
            for (int j = 0; j < 8; ++j) { afrag[mi][j] = lo[j]; afrag[mi][8 + j] = hi[j]; }
        }
        // ---- B fragments: 16-bit B 32x16 layout ----
        // lane holds column col, K = half*16 .. half*16+15 (contiguous in Bt)
        v16h bfrag[4];
#pragma unroll
        for (int ni = 0; ni < 4; ++ni)
            bfrag[ni] = *(const v16h*)(bptr[ni] + k0);
#pragma unroll
        for (int mi = 0; mi < 2; ++mi)
#pragma unroll
            for (int ni = 0; ni < 4; ++ni)
                acc[mi][ni] = __builtin_amdgcn_wmma_f32_16x16x32_f16(
                    false, afrag[mi], false, bfrag[ni],
                    (short)0, acc[mi][ni], false, false);
    }

    // ---- epilogue: C/D 16x16 f32 layout (VGPR r -> M = r + 8*half_id) ----
#pragma unroll
    for (int mi = 0; mi < 2; ++mi) {
#pragma unroll
        for (int ni = 0; ni < 4; ++ni) {
            int col = baseCol + ni * 16 + l16;
            if (col < N) {
#pragma unroll
                for (int r = 0; r < 8; ++r) {
                    int row = baseRow + mi * 16 + half_id * 8 + r;
                    size_t o = (size_t)row * N + col;
                    float v = acc[mi][ni][r];
                    if (HASADD) v += addsrc[o];
                    C[o] = v;
                }
            }
        }
    }
}

// ---------------------------------------------------------------------------
// Depthwise conv (window 4, left-padded) + SiLU on the xBC slice of zxbcdt
// ---------------------------------------------------------------------------
__global__ void k_conv_silu(const float* __restrict__ zx, const float* __restrict__ convW,
                            const float* __restrict__ convB, float* __restrict__ xBC) {
    size_t idx = (size_t)blockIdx.x * blockDim.x + threadIdx.x;
    size_t total = (size_t)MROWS * CONVDIM;
    if (idx >= total) return;
    int c   = (int)(idx % CONVDIM);
    int row = (int)(idx / CONVDIM);
    int t  = row % LL;
    int bb = row / LL;
    float s = convB[c];
#pragma unroll
    for (int k = 0; k < DCONV; ++k) {
        int tt = t - (DCONV - 1) + k;
        if (tt >= 0)
            s += zx[((size_t)(bb * LL + tt)) * DPROJ + DIN + c] * convW[k * CONVDIM + c];
    }
    xBC[idx] = silu_f(s);
}

// ---------------------------------------------------------------------------
// dt = softplus(dt_raw + dt_bias); dA = exp(dt * -exp(A_log))
// ---------------------------------------------------------------------------
__global__ void k_dt_dA(const float* __restrict__ zx, const float* __restrict__ dt_bias,
                        const float* __restrict__ A_log,
                        float* __restrict__ dt, float* __restrict__ dA) {
    int idx = blockIdx.x * blockDim.x + threadIdx.x;
    if (idx >= MROWS * NH) return;
    int h   = idx % NH;
    int row = idx / NH;
    float dtv = softplus_f(zx[(size_t)row * DPROJ + (DPROJ - NH) + h] + dt_bias[h]);
    dt[idx] = dtv;
    dA[idx] = __expf(dtv * (-__expf(A_log[h])));
}

// ---------------------------------------------------------------------------
// Selective scan: one block per (batch, head); 64 threads, thread p keeps
// state[p, 0:128] fully in registers. B/C staged through LDS each step.
// ---------------------------------------------------------------------------
__global__ __launch_bounds__(64)
void k_scan(const float* __restrict__ xBC, const float* __restrict__ dt,
            const float* __restrict__ dA, const float* __restrict__ Dv,
            float* __restrict__ y) {
    __shared__ float Bs[DSTATE];
    __shared__ float Cs[DSTATE];
    const int bh = blockIdx.x;
    const int bb = bh / NH;
    const int h  = bh % NH;
    const int p  = threadIdx.x;       // 0..63
    const float Dh = Dv[h];

    float st[DSTATE];
#pragma unroll
    for (int n = 0; n < DSTATE; ++n) st[n] = 0.0f;

    for (int t = 0; t < LL; ++t) {
        const size_t row  = (size_t)(bb * LL + t);
        const size_t base = row * CONVDIM;
        Bs[p]      = xBC[base + DIN + p];
        Bs[p + 64] = xBC[base + DIN + 64 + p];
        Cs[p]      = xBC[base + DIN + DSTATE + p];
        Cs[p + 64] = xBC[base + DIN + DSTATE + 64 + p];
        float xp  = xBC[base + h * HDIM + p];
        float dtv = dt[row * NH + h];
        float dAv = dA[row * NH + h];
        __syncthreads();
        float coef = dtv * xp;
        float yp = 0.0f;
#pragma unroll
        for (int n = 0; n < DSTATE; ++n) {
            st[n] = st[n] * dAv + coef * Bs[n];
            yp += st[n] * Cs[n];
        }
        y[row * DIN + h * HDIM + p] = yp + Dh * xp;
        __syncthreads();
    }
}

// ---------------------------------------------------------------------------
// Gated RMSNorm over d_inner=2048: fused[:,0:2048] = rmsnorm(y*silu(z))*w  (f16)
// ---------------------------------------------------------------------------
__global__ void k_gatenorm(const float* __restrict__ y, const float* __restrict__ zx,
                           const float* __restrict__ w, half_t* __restrict__ fused) {
    __shared__ float red[256];
    int row = blockIdx.x;
    int tid = threadIdx.x;
    float g[8];
    float ss = 0.f;
#pragma unroll
    for (int j = 0; j < 8; ++j) {
        int c = tid + j * 256;
        float z = zx[(size_t)row * DPROJ + c];   // z slice = zxbcdt[..., :d_inner]
        float v = y[(size_t)row * DIN + c] * silu_f(z);
        g[j] = v;
        ss += v * v;
    }
    red[tid] = ss;
    __syncthreads();
    for (int s = 128; s > 0; s >>= 1) {
        if (tid < s) red[tid] += red[tid + s];
        __syncthreads();
    }
    float scale = rsqrtf(red[0] * (1.0f / DIN) + EPS);
#pragma unroll
    for (int j = 0; j < 8; ++j) {
        int c = tid + j * 256;
        fused[(size_t)row * DFUSE + c] = (half_t)(g[j] * scale * w[c]);
    }
}

// ---------------------------------------------------------------------------
// y_zero f32 -> f16 into fused[:, 2048:4096]
// ---------------------------------------------------------------------------
__global__ void k_copy_zero(const float* __restrict__ yz, half_t* __restrict__ fused) {
    size_t idx = (size_t)blockIdx.x * blockDim.x + threadIdx.x;
    size_t total = (size_t)MROWS * DIN;
    if (idx >= total) return;
    size_t row = idx / DIN;
    int c = (int)(idx % DIN);
    fused[row * DFUSE + DIN + c] = (half_t)yz[idx];
}

// ---------------------------------------------------------------------------
// cumsum over l of y_one -> f16 into fused[:, 4096:6144]
// one thread per channel, coalesced across channels within each timestep
// ---------------------------------------------------------------------------
__global__ void k_cumsum(const float* __restrict__ yo, half_t* __restrict__ fused) {
    int bb = blockIdx.x / (DIN / 256);
    int c  = (blockIdx.x % (DIN / 256)) * 256 + threadIdx.x;
    float acc = 0.0f;
    for (int t = 0; t < LL; ++t) {
        size_t row = (size_t)(bb * LL + t);
        acc += yo[row * DIN + c];
        fused[row * DFUSE + 2 * DIN + c] = (half_t)acc;
    }
}

// ---------------------------------------------------------------------------
// launch
// ---------------------------------------------------------------------------
extern "C" void kernel_launch(void* const* d_in, const int* in_sizes, int n_in,
                              void* d_out, int out_size, void* d_ws, size_t ws_size,
                              hipStream_t stream) {
    const float* x        = (const float*)d_in[0];
    const float* norm_w   = (const float*)d_in[1];
    const float* in_projW = (const float*)d_in[2];
    const float* convW    = (const float*)d_in[3];
    const float* convB    = (const float*)d_in[4];
    const float* dt_bias  = (const float*)d_in[5];
    const float* A_log    = (const float*)d_in[6];
    const float* Dv       = (const float*)d_in[7];
    const float* gnorm_w  = (const float*)d_in[8];
    const float* zeroW    = (const float*)d_in[9];
    const float* oneW     = (const float*)d_in[10];
    const float* fusionW  = (const float*)d_in[11];
    float* out = (float*)d_out;

    char* ws = (char*)d_ws;
    size_t off = 0;
    auto take = [&](size_t bytes) -> void* {
        void* p = ws + off;
        off += (bytes + 255) & ~(size_t)255;
        return p;
    };
    float*  xn     = (float*) take((size_t)MROWS * DD * 4);
    half_t* xnh    = (half_t*)take((size_t)MROWS * DD * 2);
    half_t* Wt_ip  = (half_t*)take((size_t)DPROJ * DD * 2);
    half_t* Wt_z   = (half_t*)take((size_t)DIN * DD * 2);
    half_t* Wt_o   = (half_t*)take((size_t)DIN * DD * 2);
    half_t* Wt_f   = (half_t*)take((size_t)DD * DFUSE * 2);
    float*  zx     = (float*) take((size_t)MROWS * DPROJ * 4);
    float*  xBC    = (float*) take((size_t)MROWS * CONVDIM * 4);
    float*  dtb    = (float*) take((size_t)MROWS * NH * 4);
    float*  dAb    = (float*) take((size_t)MROWS * NH * 4);
    float*  ybuf   = (float*) take((size_t)MROWS * DIN * 4);
    float*  yz     = (float*) take((size_t)MROWS * DIN * 4);
    float*  yo     = (float*) take((size_t)MROWS * DIN * 4);
    half_t* fused  = (half_t*)take((size_t)MROWS * DFUSE * 2);

    const int T = 256;
    auto cdiv = [](size_t a, size_t b) { return (int)((a + b - 1) / b); };

    // 1) weight transposes (fp32 -> f16, N-major)
    k_transpose_f16<<<cdiv((size_t)DD * DPROJ, T), T, 0, stream>>>(in_projW, Wt_ip, DD, DPROJ);
    k_transpose_f16<<<cdiv((size_t)DD * DIN,   T), T, 0, stream>>>(zeroW,    Wt_z,  DD, DIN);
    k_transpose_f16<<<cdiv((size_t)DD * DIN,   T), T, 0, stream>>>(oneW,     Wt_o,  DD, DIN);
    k_transpose_f16<<<cdiv((size_t)DFUSE * DD, T), T, 0, stream>>>(fusionW,  Wt_f,  DFUSE, DD);

    // 2) input rmsnorm
    k_rmsnorm<<<MROWS, T, 0, stream>>>(x, norm_w, xn, xnh);

    // 3) in_proj GEMM: zx = xn @ in_proj_W   (8192 x 1024 x 4384)
    {
        dim3 g(cdiv(DPROJ, 128), MROWS / 128);
        k_gemm_wmma<false><<<g, T, 0, stream>>>(xnh, Wt_ip, zx, nullptr, MROWS, DPROJ, DD);
    }

    // 4) conv + silu, dt/dA
    k_conv_silu<<<cdiv((size_t)MROWS * CONVDIM, T), T, 0, stream>>>(zx, convW, convB, xBC);
    k_dt_dA<<<cdiv((size_t)MROWS * NH, T), T, 0, stream>>>(zx, dt_bias, A_log, dtb, dAb);

    // 5) selective scan
    k_scan<<<BB * NH, 64, 0, stream>>>(xBC, dtb, dAb, Dv, ybuf);

    // 6) y_zero / y_one GEMMs (8192 x 1024 x 2048)
    {
        dim3 g(cdiv(DIN, 128), MROWS / 128);
        k_gemm_wmma<false><<<g, T, 0, stream>>>(xnh, Wt_z, yz, nullptr, MROWS, DIN, DD);
        k_gemm_wmma<false><<<g, T, 0, stream>>>(xnh, Wt_o, yo, nullptr, MROWS, DIN, DD);
    }

    // 7) assemble fused activation (f16): [gated-norm mamba | y_zero | cumsum(y_one)]
    k_gatenorm<<<MROWS, T, 0, stream>>>(ybuf, zx, gnorm_w, fused);
    k_copy_zero<<<cdiv((size_t)MROWS * DIN, T), T, 0, stream>>>(yz, fused);
    k_cumsum<<<BB * (DIN / 256), T, 0, stream>>>(yo, fused);

    // 8) final GEMM with residual: out = x + fused @ fusion_W  (8192 x 6144 x 1024)
    {
        dim3 g(cdiv(DD, 128), MROWS / 128);
        k_gemm_wmma<true><<<g, T, 0, stream>>>(fused, Wt_f, out, x, MROWS, DD, DFUSE);
    }
}
